// PGAT_19688130085868
// MI455X (gfx1250) — compile-verified, matrix operand order checked by reading, not run on hardware
//
#include <hip/hip_runtime.h>
#include <hip/hip_bf16.h>

// ---------------- problem constants ----------------
#define NN     30000
#define EE     480000
#define IN_DIM 128
#define HID    64
#define HEADS  4
#define DOUT   256        // HEADS * HID == HEADS * OUT
#define POSD   32
#define NEG_SLOPE 0.2f

typedef __attribute__((ext_vector_type(16))) __bf16 v16bf;
typedef __attribute__((ext_vector_type(8)))  float  v8f;

// ---------------- helpers ----------------
__device__ __forceinline__ unsigned f2bf_rne(float x) {
    unsigned u = __float_as_uint(x);
    return (u + 0x7FFFu + ((u >> 16) & 1u)) >> 16;   // round-to-nearest-even bf16
}

// ---------------- fill ----------------
__global__ __launch_bounds__(256)
void fill_u32_kernel(unsigned* __restrict__ p, unsigned val, int n) {
    int i = blockIdx.x * 256 + threadIdx.x;
    if (i < n) p[i] = val;
}

// ---------------- pack W (f32 [din,256]) -> bf16 K-pair packed [din/2][256] ----------------
__global__ __launch_bounds__(256)
void pack_w_kernel(const float* __restrict__ W, unsigned* __restrict__ Wp, int din) {
    int idx = blockIdx.x * 256 + threadIdx.x;
    int total = (din >> 1) * DOUT;
    if (idx >= total) return;
    int n  = idx & (DOUT - 1);
    int kp = idx >> 8;
    unsigned lo = f2bf_rne(W[(size_t)(2 * kp)     * DOUT + n]);
    unsigned hi = f2bf_rne(W[(size_t)(2 * kp + 1) * DOUT + n]);
    Wp[idx] = lo | (hi << 16);
}

// ---------------- build hcat = concat(h, pe[pos]) in bf16 ----------------
__global__ __launch_bounds__(256)
void build_hcat_kernel(const float* __restrict__ h, int dh,
                       const float* __restrict__ pe, const int* __restrict__ pos,
                       unsigned short* __restrict__ out, int din) {
    int idx = blockIdx.x * 256 + threadIdx.x;
    if (idx >= NN * din) return;
    int n = idx / din;
    int c = idx - n * din;
    float x = (c < dh) ? h[(size_t)n * dh + c]
                       : pe[pos[n] * POSD + (c - dh)];
    out[idx] = (unsigned short)f2bf_rne(x);
}

// ---------------- WMMA GEMM: ft[N,256] = hcat[N,din](bf16) @ Wp(bf16 packed) ----------------
// 1 block = 128 thr = 4 waves; wave w computes rows [16*blockIdx.x,+16) x cols [64*w,+64)
__global__ __launch_bounds__(128)
void gemm_bf16_wmma_kernel(const unsigned short* __restrict__ A,  // [N,din] bf16
                           const unsigned* __restrict__ Wp,       // [din/2,256] packed pairs
                           float* __restrict__ C,                 // [N,256] f32
                           int din) {
    const int wave = threadIdx.x >> 5;
    const int lane = threadIdx.x & 31;
    const int g    = lane >> 4;        // half-wave group
    const int mn   = lane & 15;        // row (A) / col (B,C) within tile
    const int row0 = blockIdx.x * 16;
    const int col0 = wave * 64;

    v8f c0 = {}, c1 = {}, c2 = {}, c3 = {};
    const unsigned* arow = (const unsigned*)(A + (size_t)(row0 + mn) * din);

    for (int kb = 0; kb < din; kb += 32) {
        union { unsigned u[8]; v16bf v; } af;
#pragma unroll
        for (int v = 0; v < 8; ++v) {
            // ISA 16-bit A layout: VGPR v holds K pair (v&3)*2 (+16 for v>=4) (+8 for upper half-wave)
            int k0 = kb + ((v & 4) << 2) + ((v & 3) << 1) + (g << 3);
            af.u[v] = arow[k0 >> 1];
        }
        union { unsigned u[8]; v16bf v; } bf0, bf1, bf2, bf3;
#pragma unroll
        for (int v = 0; v < 8; ++v) {
            int k0 = kb + ((v & 4) << 2) + ((v & 3) << 1) + (g << 3);
            const unsigned* wrow = Wp + (size_t)(k0 >> 1) * DOUT + col0 + mn;
            bf0.u[v] = wrow[0];
            bf1.u[v] = wrow[16];
            bf2.u[v] = wrow[32];
            bf3.u[v] = wrow[48];
        }
        c0 = __builtin_amdgcn_wmma_f32_16x16x32_bf16(false, af.v, false, bf0.v, (short)0, c0, false, false);
        c1 = __builtin_amdgcn_wmma_f32_16x16x32_bf16(false, af.v, false, bf1.v, (short)0, c1, false, false);
        c2 = __builtin_amdgcn_wmma_f32_16x16x32_bf16(false, af.v, false, bf2.v, (short)0, c2, false, false);
        c3 = __builtin_amdgcn_wmma_f32_16x16x32_bf16(false, af.v, false, bf3.v, (short)0, c3, false, false);
    }
    // ISA f32 C layout: VGPR r, lane -> (M = r + 8*g, N = mn)
    float* crow = C + (size_t)row0 * DOUT;
#pragma unroll
    for (int r = 0; r < 8; ++r) {
        int m = r + (g << 3);
        float* p = crow + (size_t)m * DOUT + col0 + mn;
        p[0]  = c0[r];
        p[16] = c1[r];
        p[32] = c2[r];
        p[48] = c3[r];
    }
}

// ---------------- attention coefficients a1/a2 [N,H] ----------------
__global__ __launch_bounds__(256)
void attn_coeff_kernel(const float* __restrict__ ft,
                       const float* __restrict__ al, const float* __restrict__ ar,
                       float* __restrict__ a1, float* __restrict__ a2) {
    int idx = blockIdx.x * 256 + threadIdx.x;     // N*H
    if (idx >= NN * HEADS) return;
    int n  = idx >> 2;
    int hh = idx & 3;
    const float* f = ft + (size_t)n * DOUT + hh * HID;
    float s1 = 0.f, s2 = 0.f;
#pragma unroll 8
    for (int d = 0; d < HID; ++d) {
        float v = f[d];
        s1 += v * al[hh * HID + d];
        s2 += v * ar[hh * HID + d];
    }
    a1[idx] = s1;
    a2[idx] = s2;
}

// ---------------- edge pass 1: leaky-relu logits + segment max (ordered-uint atomicMax) ------
__device__ __forceinline__ unsigned enc_f32(float x) {
    unsigned u = __float_as_uint(x);
    return (u & 0x80000000u) ? ~u : (u | 0x80000000u);
}
__device__ __forceinline__ float dec_f32(unsigned u) {
    return (u & 0x80000000u) ? __uint_as_float(u & 0x7FFFFFFFu) : __uint_as_float(~u);
}

__global__ __launch_bounds__(256)
void edge_logits_kernel(const float* __restrict__ a1, const float* __restrict__ a2,
                        const int* __restrict__ src, const int* __restrict__ dst,
                        float* __restrict__ ebuf, unsigned* __restrict__ menc) {
    int idx = blockIdx.x * 256 + threadIdx.x;     // E*H
    if (idx >= EE * HEADS) return;
    int e  = idx >> 2;
    int hh = idx & 3;
    float x = a1[src[e] * HEADS + hh] + a2[dst[e] * HEADS + hh];
    x = (x > 0.f) ? x : NEG_SLOPE * x;
    ebuf[idx] = x;
    atomicMax(&menc[dst[e] * HEADS + hh], enc_f32(x));
}

// ---------------- edge pass 2: exp + segment sum ----------------
__global__ __launch_bounds__(256)
void edge_exp_kernel(const int* __restrict__ dst,
                     const unsigned* __restrict__ menc,
                     float* __restrict__ ebuf, float* __restrict__ ssum) {
    int idx = blockIdx.x * 256 + threadIdx.x;     // E*H
    if (idx >= EE * HEADS) return;
    int e  = idx >> 2;
    int hh = idx & 3;
    float m  = dec_f32(menc[dst[e] * HEADS + hh]);
    float ex = __expf(ebuf[idx] - m);
    ebuf[idx] = ex;
    atomicAdd(&ssum[dst[e] * HEADS + hh], ex);
}

// ---------------- edge pass 3: gather ft[src], scale by alpha, scatter-add to out[dst] -------
__global__ __launch_bounds__(256)
void edge_agg_kernel(const float* __restrict__ ft,
                     const float* __restrict__ ebuf, const float* __restrict__ ssum,
                     const int* __restrict__ src, const int* __restrict__ dst,
                     float* __restrict__ outacc) {
    int idx = blockIdx.x * 256 + threadIdx.x;     // E * 256 (one per edge-channel)
    int e = idx >> 8;
    if (e >= EE) return;
    int c  = idx & 255;
    int hh = c >> 6;
    int d  = dst[e];
    float alpha = ebuf[e * HEADS + hh] / ssum[d * HEADS + hh];
    float v = ft[(size_t)src[e] * DOUT + c] * alpha;
    atomicAdd(&outacc[(size_t)d * DOUT + c], v);
}

// ---------------- ELU ----------------
__global__ __launch_bounds__(256)
void elu_kernel(const float* __restrict__ in, float* __restrict__ out, int n) {
    int i = blockIdx.x * 256 + threadIdx.x;
    if (i >= n) return;
    float x = in[i];
    out[i] = (x > 0.f) ? x : (__expf(x) - 1.f);
}

// ---------------- head mean -> d_out [N,64] ----------------
__global__ __launch_bounds__(256)
void head_mean_kernel(const float* __restrict__ in, float* __restrict__ out) {
    int i = blockIdx.x * 256 + threadIdx.x;       // N*64
    if (i >= NN * HID) return;
    int n = i >> 6;
    int d = i & 63;
    const float* p = in + (size_t)n * DOUT + d;
    out[i] = 0.25f * (p[0] + p[64] + p[128] + p[192]);
}

// ==================== host launch ====================
extern "C" void kernel_launch(void* const* d_in, const int* in_sizes, int n_in,
                              void* d_out, int out_size, void* d_ws, size_t ws_size,
                              hipStream_t stream) {
    (void)in_sizes; (void)n_in; (void)out_size; (void)ws_size;

    const float* features = (const float*)d_in[0];
    const float* W[3]  = { (const float*)d_in[1], (const float*)d_in[5], (const float*)d_in[9]  };
    const float* al[3] = { (const float*)d_in[2], (const float*)d_in[6], (const float*)d_in[10] };
    const float* ar[3] = { (const float*)d_in[3], (const float*)d_in[7], (const float*)d_in[11] };
    const float* pe[3] = { (const float*)d_in[4], (const float*)d_in[8], (const float*)d_in[12] };
    const int* src = (const int*)d_in[13];
    const int* dst = (const int*)d_in[14];
    const int* pos = (const int*)d_in[15];
    float* out_final = (float*)d_out;

    const int din_l[3] = { IN_DIM + POSD, DOUT + POSD, DOUT + POSD }; // 160, 288, 288

    // ---- workspace layout ----
    char* ws = (char*)d_ws;
    size_t off = 0;
    auto take = [&](size_t bytes) -> void* {
        void* p = ws + off;
        off = (off + bytes + 255) & ~(size_t)255;
        return p;
    };
    unsigned short* hcatB = (unsigned short*)take((size_t)NN * 288 * sizeof(unsigned short));
    float*    ft     = (float*)   take((size_t)NN * DOUT * sizeof(float));
    float*    hbuf   = (float*)   take((size_t)NN * DOUT * sizeof(float));
    float*    outacc = (float*)   take((size_t)NN * DOUT * sizeof(float));
    float*    a1     = (float*)   take((size_t)NN * HEADS * sizeof(float));
    float*    a2     = (float*)   take((size_t)NN * HEADS * sizeof(float));
    float*    ebuf   = (float*)   take((size_t)EE * HEADS * sizeof(float));
    unsigned* menc   = (unsigned*)take((size_t)NN * HEADS * sizeof(unsigned));
    float*    ssum   = (float*)   take((size_t)NN * HEADS * sizeof(float));
    unsigned* Wp[3];
    for (int l = 0; l < 3; ++l)
        Wp[l] = (unsigned*)take((size_t)(din_l[l] / 2) * DOUT * sizeof(unsigned));

    auto nblk = [](long long n) { return (int)((n + 255) / 256); };

    // ---- pack weights (once per launch, deterministic) ----
    for (int l = 0; l < 3; ++l)
        pack_w_kernel<<<nblk((long long)(din_l[l] / 2) * DOUT), 256, 0, stream>>>(W[l], Wp[l], din_l[l]);

    // ---- layers ----
    for (int l = 0; l < 3; ++l) {
        const int din = din_l[l];
        const float* hin = (l == 0) ? features : hbuf;
        const int dh = (l == 0) ? IN_DIM : DOUT;

        build_hcat_kernel<<<nblk((long long)NN * din), 256, 0, stream>>>(hin, dh, pe[l], pos, hcatB, din);

        gemm_bf16_wmma_kernel<<<NN / 16, 128, 0, stream>>>(hcatB, Wp[l], ft, din);

        attn_coeff_kernel<<<nblk((long long)NN * HEADS), 256, 0, stream>>>(ft, al[l], ar[l], a1, a2);

        fill_u32_kernel<<<nblk((long long)NN * HEADS), 256, 0, stream>>>(menc, 0u, NN * HEADS);
        fill_u32_kernel<<<nblk((long long)NN * HEADS), 256, 0, stream>>>((unsigned*)ssum, 0u, NN * HEADS);
        fill_u32_kernel<<<nblk((long long)NN * DOUT), 256, 0, stream>>>((unsigned*)outacc, 0u, NN * DOUT);

        edge_logits_kernel<<<nblk((long long)EE * HEADS), 256, 0, stream>>>(a1, a2, src, dst, ebuf, menc);
        edge_exp_kernel<<<nblk((long long)EE * HEADS), 256, 0, stream>>>(dst, menc, ebuf, ssum);
        edge_agg_kernel<<<nblk((long long)EE * DOUT), 256, 0, stream>>>(ft, ebuf, ssum, src, dst, outacc);

        if (l < 2) {
            elu_kernel<<<nblk((long long)NN * DOUT), 256, 0, stream>>>(outacc, hbuf, NN * DOUT);
        } else {
            head_mean_kernel<<<nblk((long long)NN * HID), 256, 0, stream>>>(outacc, out_final);
        }
    }
}
